// Matchmaker_v1c_42674795053894
// MI455X (gfx1250) — compile-verified, matrix operand order checked by reading, not run on hardware
//
#include <hip/hip_runtime.h>
#include <hip/hip_bf16.h>

#define BATCH 128
#define QLEN  32
#define DLEN  512
#define EMB   300
#define EPAD  320
#define CH    128
#define NK    11

typedef __attribute__((ext_vector_type(16))) __bf16          v16bf;
typedef __attribute__((ext_vector_type(16))) unsigned short  v16u;
typedef __attribute__((ext_vector_type(8)))  float           v8f;
typedef __attribute__((ext_vector_type(4)))  float           f4v;
typedef __attribute__((ext_vector_type(4)))  unsigned int    u4v;
typedef __attribute__((ext_vector_type(8)))  int             i8v;
typedef __attribute__((ext_vector_type(4)))  int             i4v;

struct U256 { u4v lo, hi; };   // 32 bytes -> one v16bf fragment

#if defined(__has_builtin)
#  if __has_builtin(__builtin_amdgcn_tensor_load_to_lds) && \
      __has_builtin(__builtin_amdgcn_s_wait_tensorcnt)
#    define HAVE_TDM 1
#  endif
#endif
#ifndef HAVE_TDM
#  define HAVE_TDM 0
#endif

// RBF kernel constants (n=11): bs = 2/10 = 0.2
__device__ __constant__ float c_mu[NK]   = {1.0f, 0.9f, 0.7f, 0.5f, 0.3f, 0.1f,
                                            -0.1f, -0.3f, -0.5f, -0.7f, -0.9f};
__device__ __constant__ float c_i2s2[NK] = {500000.0f, 50.0f, 50.0f, 50.0f, 50.0f, 50.0f,
                                            50.0f, 50.0f, 50.0f, 50.0f, 50.0f};

__device__ __forceinline__ unsigned short f2bf(float f) {
    unsigned int u = __float_as_uint(f);
    unsigned int r = (u + 0x7FFFu + ((u >> 16) & 1u)) >> 16;   // RNE
    return (unsigned short)r;
}

__device__ __forceinline__ v16bf load_frag_bf16(const unsigned short* base) {
    // Two contiguous 16-byte runs, 32 bytes apart start-to-start.
    U256 u;
    u.lo = *reinterpret_cast<const u4v*>(base);
    u.hi = *reinterpret_cast<const u4v*>(base + 16);
    return __builtin_bit_cast(v16bf, u);
}

#if HAVE_TDM
// TDM: async-copy one 16-row x 128-col bf16 tile (4KB) global -> LDS.
// D# per CDNA5 ISA: group0 = {count|flags, lds_addr,
// global_addr lo, global_addr hi | type=2}; group1 = dims/strides.
__device__ __forceinline__ void tdm_load_tile(const unsigned short* gsrc,
                                              unsigned lds_byte_addr) {
    const unsigned long long ga = (unsigned long long)gsrc;
    u4v g0;
    g0[0] = 1u;                                        // count=1, user mode
    g0[1] = lds_byte_addr;                             // lds_addr [63:32]
    g0[2] = (unsigned)(ga & 0xFFFFFFFFu);              // global_addr [95:64]
    g0[3] = (unsigned)((ga >> 32) & 0x01FFFFFFu)       // global_addr [120:96]
          | 0x80000000u;                               // type=2 ("image")
    i8v g1;
    g1[0] = (int)(1u << 16);          // workgroup_mask=0, data_size=1 (2 bytes)
    g1[1] = (int)(128u << 16);        // tensor_dim0[15:0]=128 (row length)
    g1[2] = (int)(512u << 16);        // tensor_dim1[15:0]=512 (rows)
    g1[3] = (int)(128u << 16);        // tile_dim0 = 128
    g1[4] = 16;                       // tile_dim1 = 16 rows
    g1[5] = 128;                      // tensor_dim0_stride = 128 elements
    g1[6] = 0;
    g1[7] = 0;
    i4v g2 = {0, 0, 0, 0};
    i4v g3 = {0, 0, 0, 0};
#if __clang_major__ >= 23
    i8v g4 = {0, 0, 0, 0, 0, 0, 0, 0};
    __builtin_amdgcn_tensor_load_to_lds(g0, g1, g2, g3, g4, 0);
#else
    __builtin_amdgcn_tensor_load_to_lds(g0, g1, g2, g3, 0);
#endif
}
#endif

// ---------------------------------------------------------------------------
// Kernel 0: pack conv weights [C, EMB, kw] f32 -> bf16 [slot][C][EPAD]
// slot = ci*(ci+1)/2 + dk  (6 taps total), zero padded for e >= EMB.
// ---------------------------------------------------------------------------
__global__ __launch_bounds__(256)
void pack_w_kernel(const float* __restrict__ w1, const float* __restrict__ w2,
                   const float* __restrict__ w3, unsigned short* __restrict__ wp)
{
    const int gid = blockIdx.x * 256 + threadIdx.x;
    const int total = 6 * CH * EPAD;
    if (gid >= total) return;
    const int e    = gid % EPAD;
    const int c    = (gid / EPAD) % CH;
    const int slot = gid / (EPAD * CH);
    int ci, dk;
    if (slot == 0)      { ci = 0; dk = 0; }
    else if (slot <= 2) { ci = 1; dk = slot - 1; }
    else                { ci = 2; dk = slot - 3; }
    const float* w = (ci == 0) ? w1 : (ci == 1) ? w2 : w3;
    const int kw = ci + 1;
    unsigned short v = 0;
    if (e < EMB) v = f2bf(w[((size_t)c * EMB + e) * kw + dk]);
    wp[gid] = v;
}

// ---------------------------------------------------------------------------
// Kernel 1: conv (width kw = ci+1, right zero pad) + bias + ReLU + L2 norm.
// One wave computes a 16-row x 128-channel tile via WMMA bf16.
// Output: normalized bf16 rows, layout [3][B][L][CH] (ushort).
// ---------------------------------------------------------------------------
__global__ __launch_bounds__(32)
void conv_norm_kernel(const float* __restrict__ x,           // [B, L, EMB]
                      const unsigned short* __restrict__ wp, // packed weights
                      const float* __restrict__ b1, const float* __restrict__ b2,
                      const float* __restrict__ b3,
                      unsigned short* __restrict__ outn,     // [3,B,L,CH]
                      int L)
{
    const int mt = blockIdx.x;       // 16-row tile
    const int b  = blockIdx.y;
    const int ci = blockIdx.z;       // conv index 0..2
    const int kw = ci + 1;
    const float* bias = (ci == 0) ? b1 : (ci == 1) ? b2 : b3;

    const int lane = threadIdx.x;
    const int lo = lane & 15;
    const int hi = lane >> 4;

    v8f acc[8];
    #pragma unroll
    for (int n = 0; n < 8; ++n)
        #pragma unroll
        for (int r = 0; r < 8; ++r) acc[n][r] = 0.0f;

    const float* xb = x + (size_t)b * L * EMB;
    const int trow = mt * 16 + lo;   // my A-fragment row

    auto mma_tiles = [&](v16bf a, int et, int dk) {
        const int slot = (ci * (ci + 1)) / 2 + dk;
        #pragma unroll
        for (int n = 0; n < 8; ++n) {
            const int c = n * 16 + lo;
            const unsigned short* base =
                wp + (size_t)(slot * CH + c) * EPAD + et * 32 + hi * 8;
            const v16bf bf = load_frag_bf16(base);
            acc[n] = __builtin_amdgcn_wmma_f32_16x16x32_bf16(
                         false, a, false, bf, (short)0, acc[n], false, false);
        }
    };

    for (int dk = 0; dk < kw; ++dk) {
        const int srow   = trow + dk;
        const int srow_c = (srow < L - 1) ? srow : (L - 1);
        const float scale = (srow < L) ? 1.0f : 0.0f;
        const float* rp = xb + (size_t)srow_c * EMB;

        for (int et = 0; et < 9; ++et) {           // e tiles fully inside [0,300)
            const int e0 = et * 32 + hi * 8;
            const f4v f0 = *reinterpret_cast<const f4v*>(rp + e0);
            const f4v f1 = *reinterpret_cast<const f4v*>(rp + e0 + 4);
            const f4v f2 = *reinterpret_cast<const f4v*>(rp + e0 + 16);
            const f4v f3 = *reinterpret_cast<const f4v*>(rp + e0 + 20);
            v16u au;
            #pragma unroll
            for (int i = 0; i < 4; ++i) {
                au[i]      = f2bf(f0[i] * scale);
                au[4 + i]  = f2bf(f1[i] * scale);
                au[8 + i]  = f2bf(f2[i] * scale);
                au[12 + i] = f2bf(f3[i] * scale);
            }
            mma_tiles(__builtin_bit_cast(v16bf, au), et, dk);
        }
        {                                           // et == 9: 288..319 crosses EMB
            v16u au;
            #pragma unroll
            for (int m = 0; m < 16; ++m) {
                const int e   = 288 + ((m < 8) ? 0 : 16) + hi * 8 + (m & 7);
                const int e_c = (e < EMB) ? e : (EMB - 1);
                const float s = (e < EMB) ? scale : 0.0f;
                au[m] = f2bf(rp[e_c] * s);
            }
            mma_tiles(__builtin_bit_cast(v16bf, au), 9, dk);
        }
    }

    // bias + ReLU
    #pragma unroll
    for (int n = 0; n < 8; ++n) {
        const float bb = bias[n * 16 + lo];
        #pragma unroll
        for (int r = 0; r < 8; ++r) {
            float v = acc[n][r] + bb;
            acc[n][r] = (v > 0.0f) ? v : 0.0f;
        }
    }

    // L2 norm per output row (row = r + 8*hi): 16-lane butterfly per half-wave.
    float inv[8];
    #pragma unroll
    for (int r = 0; r < 8; ++r) {
        float s = 0.0f;
        #pragma unroll
        for (int n = 0; n < 8; ++n) s += acc[n][r] * acc[n][r];
        s += __shfl_xor(s, 1, 32);
        s += __shfl_xor(s, 2, 32);
        s += __shfl_xor(s, 4, 32);
        s += __shfl_xor(s, 8, 32);
        inv[r] = 1.0f / (sqrtf(s) + 1e-13f);
    }

    unsigned short* ob = outn + (((size_t)ci * BATCH + b) * L + mt * 16) * CH;
    #pragma unroll
    for (int n = 0; n < 8; ++n)
        #pragma unroll
        for (int r = 0; r < 8; ++r) {
            const int row = r + 8 * hi;
            ob[(size_t)row * CH + n * 16 + lo] = f2bf(acc[n][r] * inv[r]);
        }
}

// ---------------------------------------------------------------------------
// Kernel 2: cosine GEMM (M=16 q rows, N=512 docs, K=128) + RBF kernels +
// sum over doc dimension, fused per wave. Doc tiles are staged into LDS by
// the Tensor Data Mover (double-buffered, s_wait_tensorcnt synchronized);
// B fragments then come from ds_load_b128. Output pkq [9][B][QLEN][NK].
// ---------------------------------------------------------------------------
__global__ __launch_bounds__(32)
void cosrbf_kernel(const unsigned short* __restrict__ qn,  // [3,B,Q,CH] bf16
                   const unsigned short* __restrict__ dn,  // [3,B,D,CH] bf16
                   const float* __restrict__ qm,           // [B,Q]
                   const float* __restrict__ dm,           // [B,D]
                   float* __restrict__ pkq)                // [9,B,Q,NK]
{
    const int qt   = blockIdx.x;   // 0..1 (16 q rows each)
    const int b    = blockIdx.y;
    const int pair = blockIdx.z;   // 0..8 ; pair = qi*3 + dj
    const int qi = pair / 3, dj = pair % 3;

    const int lane = threadIdx.x;
    const int lo = lane & 15;
    const int hi = lane >> 4;

    const unsigned short* qb = qn + (((size_t)qi * BATCH + b) * QLEN + qt * 16) * CH;
    const unsigned short* db = dn + (((size_t)dj * BATCH + b) * DLEN) * CH;

#if HAVE_TDM
    __shared__ unsigned short sbuf[2][16 * CH];    // 2 x 4KB doc tiles
    const unsigned lds0 = (unsigned)(unsigned long long)&sbuf[0][0];
    const unsigned lds1 = (unsigned)(unsigned long long)&sbuf[1][0];
    tdm_load_tile(db, lds0);                       // prefetch tile 0
#endif

    // A fragments: 16 q rows x 128 channels -> 4 K-tiles, loaded once (b128s).
    v16bf aF[4];
    #pragma unroll
    for (int kt = 0; kt < 4; ++kt)
        aF[kt] = load_frag_bf16(qb + (size_t)lo * CH + kt * 32 + hi * 8);

    float qmv[8];
    #pragma unroll
    for (int r = 0; r < 8; ++r) qmv[r] = qm[b * QLEN + qt * 16 + r + 8 * hi];

    float sums[8][NK];
    #pragma unroll
    for (int r = 0; r < 8; ++r)
        #pragma unroll
        for (int k = 0; k < NK; ++k) sums[r][k] = 0.0f;

    for (int nt = 0; nt < 32; ++nt) {              // 32 doc tiles of 16
        v8f acc;
        #pragma unroll
        for (int r = 0; r < 8; ++r) acc[r] = 0.0f;

        const int d = nt * 16 + lo;                // B lane column = doc index

#if HAVE_TDM
        if (nt + 1 < 32) {                         // issue next tile, then wait
            tdm_load_tile(db + (size_t)(nt + 1) * 16 * CH,
                          ((nt + 1) & 1) ? lds1 : lds0);
            __builtin_amdgcn_s_wait_tensorcnt(1);  // tile nt landed
        } else {
            __builtin_amdgcn_s_wait_tensorcnt(0);
        }
        const unsigned short* tile = &sbuf[nt & 1][0];
        #pragma unroll
        for (int kt = 0; kt < 4; ++kt) {
            const v16bf bf = load_frag_bf16(tile + (size_t)lo * CH + kt * 32 + hi * 8);
            acc = __builtin_amdgcn_wmma_f32_16x16x32_bf16(
                      false, aF[kt], false, bf, (short)0, acc, false, false);
        }
#else
        #pragma unroll
        for (int kt = 0; kt < 4; ++kt) {
            const v16bf bf = load_frag_bf16(db + (size_t)d * CH + kt * 32 + hi * 8);
            acc = __builtin_amdgcn_wmma_f32_16x16x32_bf16(
                      false, aF[kt], false, bf, (short)0, acc, false, false);
        }
#endif

        const float dmv = dm[b * DLEN + d];
        #pragma unroll
        for (int r = 0; r < 8; ++r) {
            const float m  = qmv[r] * dmv;
            const float cm = acc[r] * m;           // masked cosine
            #pragma unroll
            for (int k = 0; k < NK; ++k) {
                const float dlt = cm - c_mu[k];
                sums[r][k] += __expf(-dlt * dlt * c_i2s2[k]) * m;
            }
        }
    }

    // Reduce the doc-sum across the 16 lanes of each half-wave.
    #pragma unroll
    for (int r = 0; r < 8; ++r)
        #pragma unroll
        for (int k = 0; k < NK; ++k) {
            float v = sums[r][k];
            v += __shfl_xor(v, 1, 32);
            v += __shfl_xor(v, 2, 32);
            v += __shfl_xor(v, 4, 32);
            v += __shfl_xor(v, 8, 32);
            sums[r][k] = v;
        }

    if (lo == 0) {                                 // lanes 0 and 16 store
        float* o = pkq + (((size_t)pair * BATCH + b) * QLEN + qt * 16 + 8 * hi) * NK;
        #pragma unroll
        for (int r = 0; r < 8; ++r)
            #pragma unroll
            for (int k = 0; k < NK; ++k)
                o[r * NK + k] = sums[r][k];
    }
}

// ---------------------------------------------------------------------------
// Kernel 3: log-pool over q, dense heads, combination. One block per batch.
// ---------------------------------------------------------------------------
__global__ __launch_bounds__(128)
void final_kernel(const float* __restrict__ pkq, const float* __restrict__ qm,
                  const float* __restrict__ dm, const float* __restrict__ ns,
                  const float* __restrict__ dw, const float* __restrict__ dbv,
                  const float* __restrict__ dmw, const float* __restrict__ dmb,
                  const float* __restrict__ dcw, float* __restrict__ out)
{
    const int b = blockIdx.x;
    const int tid = threadIdx.x;
    __shared__ float red[128];

    float dl = 0.0f;                               // doc_len = sum(dm[b,:])
    for (int d = tid; d < DLEN; d += 128) dl += dm[b * DLEN + d];
    red[tid] = dl; __syncthreads();
    for (int s = 64; s > 0; s >>= 1) {
        if (tid < s) red[tid] += red[tid + s];
        __syncthreads();
    }
    dl = red[0]; __syncthreads();
    const float invdl = 1.0f / dl;
    const float scaler = ns[0];

    float s1 = 0.0f, s2 = 0.0f;
    for (int idx = tid; idx < 9 * QLEN * NK; idx += 128) {
        const int k    = idx % NK;
        const int rem  = idx / NK;
        const int q    = rem % QLEN;
        const int pair = rem / QLEN;
        const float v  = pkq[(((size_t)pair * BATCH + b) * QLEN + q) * NK + k];
        const float qq = qm[b * QLEN + q] * scaler;
        const int f    = pair * NK + k;
        s1 += logf(fmaxf(v,         1e-10f)) * qq * dw[f];
        s2 += logf(fmaxf(v * invdl, 1e-10f)) * qq * dmw[f];
    }

    red[tid] = s1; __syncthreads();
    for (int s = 64; s > 0; s >>= 1) {
        if (tid < s) red[tid] += red[tid + s];
        __syncthreads();
    }
    s1 = red[0]; __syncthreads();

    red[tid] = s2; __syncthreads();
    for (int s = 64; s > 0; s >>= 1) {
        if (tid < s) red[tid] += red[tid + s];
        __syncthreads();
    }
    s2 = red[0];

    if (tid == 0)
        out[b] = (s1 + dbv[0]) * dcw[0] + (s2 + dmb[0]) * dcw[1];
}

// ---------------------------------------------------------------------------
extern "C" void kernel_launch(void* const* d_in, const int* in_sizes, int n_in,
                              void* d_out, int out_size, void* d_ws, size_t ws_size,
                              hipStream_t stream)
{
    const float* qe  = (const float*)d_in[0];
    const float* de  = (const float*)d_in[1];
    const float* qm  = (const float*)d_in[2];
    const float* dm  = (const float*)d_in[3];
    const float* c1w = (const float*)d_in[4];
    const float* c1b = (const float*)d_in[5];
    const float* c2w = (const float*)d_in[6];
    const float* c2b = (const float*)d_in[7];
    const float* c3w = (const float*)d_in[8];
    const float* c3b = (const float*)d_in[9];
    const float* ns  = (const float*)d_in[10];
    const float* dw  = (const float*)d_in[11];
    const float* dbv = (const float*)d_in[12];
    const float* dmw = (const float*)d_in[13];
    const float* dmb = (const float*)d_in[14];
    const float* dcw = (const float*)d_in[15];
    float* out = (float*)d_out;

    char* ws = (char*)d_ws;
    const size_t qn_bytes = (size_t)3 * BATCH * QLEN * CH * sizeof(unsigned short);
    const size_t dn_bytes = (size_t)3 * BATCH * DLEN * CH * sizeof(unsigned short);
    const size_t pk_bytes = (size_t)9 * BATCH * QLEN * NK * sizeof(float);
    unsigned short* qn = (unsigned short*)ws;
    unsigned short* dn = (unsigned short*)(ws + qn_bytes);
    float*         pkq = (float*)(ws + qn_bytes + dn_bytes);
    unsigned short* wp = (unsigned short*)(ws + qn_bytes + dn_bytes + pk_bytes);

    const int wtotal = 6 * CH * EPAD;
    pack_w_kernel<<<dim3((wtotal + 255) / 256), 256, 0, stream>>>(c1w, c2w, c3w, wp);
    conv_norm_kernel<<<dim3(QLEN / 16, BATCH, 3), 32, 0, stream>>>(
        qe, wp, c1b, c2b, c3b, qn, QLEN);
    conv_norm_kernel<<<dim3(DLEN / 16, BATCH, 3), 32, 0, stream>>>(
        de, wp, c1b, c2b, c3b, dn, DLEN);
    cosrbf_kernel<<<dim3(2, BATCH, 9), 32, 0, stream>>>(qn, dn, qm, dm, pkq);
    final_kernel<<<dim3(BATCH), 128, 0, stream>>>(
        pkq, qm, dm, ns, dw, dbv, dmw, dmb, dcw, out);
}
